// ParagraphVector_90039694393585
// MI455X (gfx1250) — matrix operand order repaired and checked by reading, not compiled, not run on hardware
//
#include <hip/hip_runtime.h>

// ---------------------------------------------------------------------------
// ParagraphVector fwd.  Gather-bound (~280MB traffic, 0.15 GFLOP @ 23.3TB/s
// => ~10us memory floor), so: wave-per-b, coalesced float4 gathers, f16
// staged in LDS, 72 length-128 dots on the matrix pipe as 5 chained-K
// v_wmma_f32_16x16x32_f16 tiles against a column-broadcast B.  All hot-path
// control flow is compile-time (full tile unroll) or single cndmasks.
// ---------------------------------------------------------------------------

#define B_N   8192
#define W_N   8
#define E_N   128
#define NNEG  8
#define KROWS (W_N * NNEG)        // 64 neg rows
#define ROWS  (W_N + KROWS)       // 72 real rows, padded to 80 = 5x16
#define TILES 5
#define WAVES_PER_BLOCK 8
#define THREADS (WAVES_PER_BLOCK * 32)
#define TILE_H_STRIDE 136         // halfs per staged row (272B, bank-spread)

typedef __attribute__((ext_vector_type(16))) _Float16 v16h;
typedef __attribute__((ext_vector_type(8)))  _Float16 v8h;
typedef __attribute__((ext_vector_type(4)))  _Float16 v4h;
typedef __attribute__((ext_vector_type(8)))  float    v8f;

struct alignas(16) WaveSmem {
    float    re[E_N];                       // review_emb[b], f32      (512B)
    _Float16 tile[16 * TILE_H_STRIDE];      // staged A tile, f16      (4352B)
    float    rowW[80];                      // per-row mask weight     (320B)
};

__device__ __forceinline__ float softplusf(float x) {
    // branchless, overflow-safe: max(x,0) + log(1 + exp(-|x|))
    return fmaxf(x, 0.f) + __logf(1.f + __expf(-fabsf(x)));
}

__global__ __launch_bounds__(THREADS)
void ParagraphVector_90039694393585_kernel(
        const int*   __restrict__ review_ids,       // [B]
        const float* __restrict__ review_word_emb,  // [B,W,E]
        const int*   __restrict__ review_word_mask, // [B,W]
        const int*   __restrict__ neg_sample_idxs,  // [B,KROWS]
        const float* __restrict__ review_emb_table, // [REVIEW_COUNT,E]
        const float* __restrict__ word_emb_table,   // [VOCAB,E]
        float*       __restrict__ out_emb,          // [B,E]
        float*       __restrict__ out_loss)         // [B]
{
    __shared__ WaveSmem smem[WAVES_PER_BLOCK];

    const int wave = threadIdx.x >> 5;
    const int lane = threadIdx.x & 31;
    const int b    = blockIdx.x * WAVES_PER_BLOCK + wave;   // wave-uniform
    if (b >= B_N) return;

    WaveSmem& ws = smem[wave];

    // ---- gather review_emb[b] (float4/lane), emit first output ----
    const int    rid = review_ids[b];
    const float4 re4 = *(const float4*)(review_emb_table + (size_t)rid * E_N + lane * 4);
    *(float4*)(ws.re + lane * 4)                     = re4;
    *(float4*)(out_emb + (size_t)b * E_N + lane * 4) = re4;

    // ---- prefetch the 64 gathered neg rows into L2 (global_prefetch_b8) ----
    const int* nidx = neg_sample_idxs + b * KROWS;
    __builtin_prefetch((const void*)(word_emb_table + (size_t)nidx[lane]      * E_N), 0, 0);
    __builtin_prefetch((const void*)(word_emb_table + (size_t)nidx[lane + 32] * E_N), 0, 0);

    // ---- per-row weight table: mask for real rows, 0 for pad rows ----
    // row r: r<8 -> pos word r; 8<=r<72 -> neg (word (r-8)/8); 72..79 pad.
    const int* mk = review_word_mask + b * W_N;
    {
        const int r0 = lane;            // 0..31
        const int r1 = lane + 32;       // 32..63
        const int r2 = lane + 64;       // 64..95 (only <80 for lane<16)
        const int w0 = (r0 < W_N) ? r0 : ((r0 - W_N) >> 3);
        ws.rowW[r0] = (float)mk[w0];
        ws.rowW[r1] = (float)mk[(r1 - W_N) >> 3];
        if (r2 < 80)
            ws.rowW[r2] = (r2 < ROWS) ? (float)mk[(r2 - W_N) >> 3] : 0.f;
    }
    // mask sum (all lanes; wave-internal LDS ops are in-order => visible)
    float msum = 0.f;
    {
        const float4 m0 = *(const float4*)(ws.rowW + 0);
        const float4 m1 = *(const float4*)(ws.rowW + 4);
        msum = (m0.x + m0.y + m0.z + m0.w) + (m1.x + m1.y + m1.z + m1.w);
    }

    // ---- broadcast-column B chunk registers (K=32 each) ----
    // B layout (32x16 f16): lane = column N (all identical here);
    // halfs 0..7 hold K = base..base+7, halfs 8..15 hold K = base+16..base+23,
    // base = 32c + (lane<16 ? 0 : 8).
    v16h Bc[4];
    const int bbase = (lane & 16) ? 8 : 0;
#pragma unroll
    for (int c = 0; c < 4; ++c) {
        const float* s = ws.re + c * 32 + bbase;
        v16h t;
#pragma unroll
        for (int j = 0; j < 16; ++j) t[j] = (_Float16)s[(j < 8) ? j : j + 8];
        Bc[c] = t;
    }

    // ---- 5 tiles: stage f16 -> 4 chained WMMA -> branchless reduce ----
    float     partial = 0.f;
    const int hi8     = (lane & 16) ? 8 : 0;   // D: VGPR v -> M = v + hi8
    const int arow    = lane & 15;             // A: lane -> row M
    const int abase   = (lane & 16) ? 8 : 0;   // A: K sub-block per half

    const _Float16* aptr = ws.tile + arow * TILE_H_STRIDE + abase;

#pragma unroll
    for (int t = 0; t < TILES; ++t) {
        // stage 16 rows: global float4 -> f16x4 (8B) per lane, per row
#pragma unroll
        for (int r = 0; r < 16; ++r) {
            const int gr = t * 16 + r;         // compile-time constant
            float4    v4 = make_float4(0.f, 0.f, 0.f, 0.f);
            if (gr < W_N) {
                v4 = *(const float4*)(review_word_emb +
                                      ((size_t)b * W_N + gr) * E_N + lane * 4);
            } else if (gr < ROWS) {
                const int wi = nidx[gr - W_N];
                v4 = *(const float4*)(word_emb_table + (size_t)wi * E_N + lane * 4);
            }
            v4h h;
            h[0] = (_Float16)v4.x; h[1] = (_Float16)v4.y;
            h[2] = (_Float16)v4.z; h[3] = (_Float16)v4.w;
            *(v4h*)(ws.tile + r * TILE_H_STRIDE + lane * 4) = h;
        }

        // K = 128 via 4 chained v_wmma_f32_16x16x32_f16
        // A layout: lane holds row M = lane&15; halfs 0..7 = K base..base+7,
        // halfs 8..15 = K base+16..base+23, base = 32c + (lane<16?0:8).
        v8f acc = {0.f, 0.f, 0.f, 0.f, 0.f, 0.f, 0.f, 0.f};
#pragma unroll
        for (int c = 0; c < 4; ++c) {
            const v8h lo = *(const v8h*)(aptr + c * 32);
            const v8h hi = *(const v8h*)(aptr + c * 32 + 16);
            const v16h a = __builtin_shufflevector(lo, hi,
                    0, 1, 2, 3, 4, 5, 6, 7, 8, 9, 10, 11, 12, 13, 14, 15);
            acc = __builtin_amdgcn_wmma_f32_16x16x32_f16(
                    false, a, false, Bc[c], (short)0, acc, false, false);
        }

        // D column 0 holds this tile's 16 dots (every lane of a half-wave
        // holds the same 8 values).  Rows handled: t*16+hi8 .. +7.
        const float4 w0 = *(const float4*)(ws.rowW + t * 16 + hi8);
        const float4 w1 = *(const float4*)(ws.rowW + t * 16 + hi8 + 4);
        const float  wv[8] = {w0.x, w0.y, w0.z, w0.w, w1.x, w1.y, w1.z, w1.w};
        // positive rows (tile 0, low half) use softplus(-x): one cndmask
        const float sgn = (t == 0) ? ((hi8 == 0) ? -1.f : 1.f) : 1.f;
#pragma unroll
        for (int v = 0; v < 8; ++v)
            partial += wv[v] * softplusf(sgn * acc[v]);
    }

    // combine the M=0..7 half with the M=8..15 half
    const float total = partial + __shfl_xor(partial, 16, 32);
    if (lane == 0)
        out_loss[b] = total / fmaxf(msum, 1.f);
}

extern "C" void kernel_launch(void* const* d_in, const int* in_sizes, int n_in,
                              void* d_out, int out_size, void* d_ws, size_t ws_size,
                              hipStream_t stream) {
    (void)in_sizes; (void)n_in; (void)out_size; (void)d_ws; (void)ws_size;

    const int*   review_ids       = (const int*)  d_in[0];
    const float* review_word_emb  = (const float*)d_in[1];
    const int*   review_word_mask = (const int*)  d_in[2];
    const int*   neg_sample_idxs  = (const int*)  d_in[3];
    /* d_in[4] = n_negs scalar (compile-time NNEG=8) */
    const float* review_emb_table = (const float*)d_in[5];
    const float* word_emb_table   = (const float*)d_in[6];

    float* out      = (float*)d_out;
    float* out_emb  = out;                       // [B, E]
    float* out_loss = out + (size_t)B_N * E_N;   // [B]

    const dim3 grid(B_N / WAVES_PER_BLOCK);
    ParagraphVector_90039694393585_kernel<<<grid, THREADS, 0, stream>>>(
        review_ids, review_word_emb, review_word_mask, neg_sample_idxs,
        review_emb_table, word_emb_table, out_emb, out_loss);
}
